// BasicConvolutionBlock_14594298871916
// MI455X (gfx1250) — matrix operand order repaired and checked by reading, not compiled, not run on hardware
//
#include <hip/hip_runtime.h>
#include <stdint.h>

#define INC   64
#define OUTC  128
#define KK    27
#define EPS   1e-5f

typedef __attribute__((ext_vector_type(16))) __bf16 bf16x16;
typedef __attribute__((ext_vector_type(8)))  float  v8f;
typedef int v4i_n __attribute__((vector_size(16)));   // native vector for builtins

union BF16Frag {
    uint4   q[2];
    bf16x16 v;
};

#if __has_builtin(__builtin_amdgcn_global_load_async_to_lds_b128)
#define HAVE_ASYNC_LDS 1
#else
#define HAVE_ASYNC_LDS 0
#endif

__device__ __forceinline__ void wait_async_all() {
#if __has_builtin(__builtin_amdgcn_s_wait_asynccnt)
    __builtin_amdgcn_s_wait_asynccnt(0);
#else
    asm volatile("s_wait_asynccnt 0x0" ::: "memory");
#endif
}

// Stage one 16KB weight tile (1024 x 16B) into LDS with 256 threads.
// Exactly 4 transfers per thread -> no exec-mask cascade.
__device__ __forceinline__ void stage_tile(const unsigned short* __restrict__ src,
                                           unsigned short* dst, int tid) {
    const uint4* s = (const uint4*)src;
    uint4* d = (uint4*)dst;
#if HAVE_ASYNC_LDS
    #pragma unroll
    for (int j = 0; j < 4; ++j) {
        int i = tid + j * 256;
        __builtin_amdgcn_global_load_async_to_lds_b128(
            (v4i_n*)(s + i), (v4i_n*)(d + i), 0, 0);
    }
#else
    #pragma unroll
    for (int j = 0; j < 4; ++j) {
        int i = tid + j * 256;
        d[i] = s[i];
    }
#endif
}

__device__ __forceinline__ unsigned short f2bf(float x) {
    uint32_t u = __float_as_uint(x);
    uint32_t r = u + 0x7FFFu + ((u >> 16) & 1u);   // round-to-nearest-even
    return (unsigned short)(r >> 16);
}

// ---------------------------------------------------------------------------
// Kernel 0: feats -> bf16; weight -> bf16 transposed [k][outc][inc]; zero stats.
// ---------------------------------------------------------------------------
__global__ void prep_kernel(const float* __restrict__ feats,
                            const float* __restrict__ weight,
                            unsigned short* __restrict__ featsB,
                            unsigned short* __restrict__ wT,
                            float* __restrict__ gstats,   // 256 floats: sum, sumsq
                            int n)
{
    const int total_f = n * INC;
    const int total_w = KK * OUTC * INC;
    int i = blockIdx.x * blockDim.x + threadIdx.x;
    int stride = gridDim.x * blockDim.x;
    for (; i < total_f; i += stride) {
        featsB[i] = f2bf(feats[i]);
        if (i < total_w) {
            int k = i >> 13;
            int r = i & 8191;
            int o = r >> 6;
            int c = r & 63;
            wT[i] = f2bf(weight[(k * INC + c) * OUTC + o]);
        }
        if (i < 256) gstats[i] = 0.0f;
    }
}

// ---------------------------------------------------------------------------
// Kernel 1: gathered conv via bf16 WMMA + per-channel partial stats.
// ---------------------------------------------------------------------------
__global__ __launch_bounds__(256, 1)
void conv_kernel(const unsigned short* __restrict__ featsB,
                 const unsigned short* __restrict__ wT,
                 const int* __restrict__ nmap,
                 float* __restrict__ out,
                 float* __restrict__ gsum,
                 float* __restrict__ gsq,
                 int n)
{
    __shared__ unsigned short wlds[2][OUTC * INC];   // 2 x 16KB double buffer
    __shared__ float s_sum[OUTC];
    __shared__ float s_sq[OUTC];

    const int tid    = threadIdx.x;
    const int wave   = tid >> 5;
    const int lane   = tid & 31;
    const int lane16 = lane & 15;
    const bool hi    = lane >= 16;
    const int h8     = hi ? 1 : 0;
    const int rowBase = blockIdx.x * 128 + wave * 16;
    const int myRow   = rowBase + lane16;          // row this lane gathers (A)

    if (tid < OUTC) { s_sum[tid] = 0.0f; s_sq[tid] = 0.0f; }

    stage_tile(wT, &wlds[0][0], tid);
    wait_async_all();
    __syncthreads();

    v8f acc[8] = {};

    for (int k = 0; k < KK; ++k) {
        if (k + 1 < KK)
            stage_tile(wT + (size_t)(k + 1) * OUTC * INC, &wlds[(k + 1) & 1][0], tid);
        if (k + 2 < KK)
            __builtin_prefetch(wT + (size_t)(k + 2) * OUTC * INC + tid * 32, 0, 1);

        int idx = (myRow < n) ? nmap[(size_t)k * n + myRow] : -1;
        const uint4* fb = (const uint4*)(featsB + (size_t)(idx < 0 ? 0 : idx) * INC);
        const unsigned short* wb = &wlds[k & 1][0];

        #pragma unroll
        for (int c = 0; c < 2; ++c) {
            // A: lane holds row M = lane16; per 32-K chunk, lo lanes take halfs
            // {0-7,16-23}, hi lanes {8-15,24-31}  (ISA 16-bit A layout).
            BF16Frag a;
            if (idx >= 0) {
                a.q[0] = fb[4 * c + h8];
                a.q[1] = fb[4 * c + h8 + 2];
            } else {
                a.q[0] = make_uint4(0, 0, 0, 0);
                a.q[1] = make_uint4(0, 0, 0, 0);
            }

            // B: lane holds column N = lane16; lo lanes K = c*32+{0-7,16-23},
            // hi lanes K = c*32+{8-15,24-31} (same order as A).
            const unsigned short* bbase = wb + lane16 * INC + c * 32 + (hi ? 8 : 0);

            BF16Frag b[2];
            b[0].q[0] = ((const uint4*)bbase)[0];
            b[0].q[1] = ((const uint4*)bbase)[2];
            #pragma unroll
            for (int nt = 0; nt < 8; ++nt) {
                if (nt < 7) {
                    const uint4* np = (const uint4*)(bbase + (nt + 1) * 16 * INC);
                    b[(nt + 1) & 1].q[0] = np[0];
                    b[(nt + 1) & 1].q[1] = np[2];
                }
                acc[nt] = __builtin_amdgcn_wmma_f32_16x16x32_bf16(
                    false, a.v, false, b[nt & 1].v, (short)0, acc[nt], false, false);
            }
        }
        wait_async_all();
        __syncthreads();   // staging of k+1 done AND all reads of buf k done
    }

    // Epilogue: C layout -> row M = v + (hi?8:0), col N = lane16 (per tile).
    #pragma unroll
    for (int nt = 0; nt < 8; ++nt) {
        const int col = nt * 16 + lane16;
        float ls = 0.0f, lq = 0.0f;
        #pragma unroll
        for (int v = 0; v < 8; ++v) {
            float x = acc[nt][v];
            ls += x;
            lq += x * x;
            int row = rowBase + v + (hi ? 8 : 0);
            if (row < n) out[(size_t)row * OUTC + col] = x;
        }
        atomicAdd(&s_sum[col], ls);   // LDS f32 atomic
        atomicAdd(&s_sq[col], lq);
    }
    __syncthreads();
    if (tid < OUTC) {
        unsafeAtomicAdd(&gsum[tid], s_sum[tid]);  // global_atomic_add_f32
        unsafeAtomicAdd(&gsq[tid],  s_sq[tid]);
    }
}

// ---------------------------------------------------------------------------
// Kernel 2: finalize per-channel scale/bias.
// ---------------------------------------------------------------------------
__global__ void finalize_kernel(const float* __restrict__ gsum,
                                const float* __restrict__ gsq,
                                const float* __restrict__ gamma,
                                const float* __restrict__ beta,
                                float* __restrict__ scale,
                                float* __restrict__ bias,
                                int n)
{
    int c = threadIdx.x;
    if (c < OUTC) {
        float invN = 1.0f / (float)n;
        float mean = gsum[c] * invN;
        float var  = gsq[c] * invN - mean * mean;
        float s = gamma[c] * rsqrtf(var + EPS);
        scale[c] = s;
        bias[c]  = beta[c] - mean * s;
    }
}

// ---------------------------------------------------------------------------
// Kernel 3: in-place norm + ReLU, float4 grid-stride. The channel group of a
// thread is grid-stride invariant (stride*4 % 128 == 0), so hoist scale/bias.
// ---------------------------------------------------------------------------
__global__ void norm_kernel(float* __restrict__ out,
                            const float* __restrict__ scale,
                            const float* __restrict__ bias,
                            int total4)
{
    float4* o = (float4*)out;
    int i = blockIdx.x * blockDim.x + threadIdx.x;
    int stride = gridDim.x * blockDim.x;
    const int c = (i << 2) & (OUTC - 1);
    const float s0 = scale[c + 0], s1 = scale[c + 1], s2 = scale[c + 2], s3 = scale[c + 3];
    const float b0 = bias[c + 0],  b1 = bias[c + 1],  b2 = bias[c + 2],  b3 = bias[c + 3];
    for (; i < total4; i += stride) {
        float4 v = o[i];
        v.x = fmaxf(v.x * s0 + b0, 0.0f);
        v.y = fmaxf(v.y * s1 + b1, 0.0f);
        v.z = fmaxf(v.z * s2 + b2, 0.0f);
        v.w = fmaxf(v.w * s3 + b3, 0.0f);
        o[i] = v;
    }
}

// ---------------------------------------------------------------------------
extern "C" void kernel_launch(void* const* d_in, const int* in_sizes, int n_in,
                              void* d_out, int out_size, void* d_ws, size_t ws_size,
                              hipStream_t stream)
{
    const float* feats  = (const float*)d_in[0];
    const float* weight = (const float*)d_in[1];
    const float* gamma  = (const float*)d_in[2];
    const float* beta   = (const float*)d_in[3];
    const int*   nmap   = (const int*)d_in[4];
    float* out = (float*)d_out;

    const int n = in_sizes[0] / INC;   // 200000

    // Workspace layout (bytes):
    char* ws = (char*)d_ws;
    unsigned short* featsB = (unsigned short*)ws;                         // n*64*2
    size_t off = (size_t)n * INC * sizeof(unsigned short);
    off = (off + 255) & ~(size_t)255;
    unsigned short* wT = (unsigned short*)(ws + off);                     // 27*128*64*2
    off += (size_t)KK * OUTC * INC * sizeof(unsigned short);
    off = (off + 255) & ~(size_t)255;
    float* gstats = (float*)(ws + off);                                   // 256 f32
    float* gsum = gstats;
    float* gsq  = gstats + OUTC;
    off += 256 * sizeof(float);
    float* scale = (float*)(ws + off);
    float* bias  = scale + OUTC;

    // 0) precision conversion + zero stats
    {
        int total = n * INC;
        int blocks = (total + 255) / 256;
        if (blocks > 8192) blocks = 8192;
        prep_kernel<<<blocks, 256, 0, stream>>>(feats, weight, featsB, wT, gstats, n);
    }
    // 1) gathered WMMA conv + stats
    {
        int blocks = (n + 127) / 128;
        conv_kernel<<<blocks, 256, 0, stream>>>(featsB, wT, nmap, out, gsum, gsq, n);
    }
    // 2) finalize scale/bias
    finalize_kernel<<<1, 128, 0, stream>>>(gsum, gsq, gamma, beta, scale, bias, n);
    // 3) norm + relu
    {
        int total4 = n * (OUTC / 4);
        int blocks = (total4 + 255) / 256;
        if (blocks > 8192) blocks = 8192;
        norm_kernel<<<blocks, 256, 0, stream>>>(out, scale, bias, total4);
    }
}
